// MultiHeadAttention_40209483825887
// MI455X (gfx1250) — matrix operand order repaired and checked by reading, not compiled, not run on hardware
//
#include <hip/hip_runtime.h>
#include <hip/hip_bf16.h>

typedef _Float16 half_t;
typedef __attribute__((ext_vector_type(16))) _Float16 v16h;
typedef __attribute__((ext_vector_type(8)))  _Float16 v8h;
typedef __attribute__((ext_vector_type(8)))  float    v8f;

#define DEVFN static __device__ __forceinline__

constexpr int Bc = 4, Sc = 1024, Dc = 1024, Hc = 16;
constexpr int Mrows = Bc * Sc;            // 4096 rows of LN'd activations
constexpr float LN_EPS = 1e-5f;
constexpr float QK_SCALE = 0.125f;        // 1/sqrt(64)

// ---------------- WMMA fragment helpers (CDNA5 16x16x32 f16 layouts) -------

DEVFN v8f wmma_f16(v16h a, v16h b, v8f c) {
  return __builtin_amdgcn_wmma_f32_16x16x32_f16(false, a, false, b,
                                                (short)0, c, false, false);
}

// A fragment 16x32 (MxK) from row-major f16, base pre-offset to (row0,k0).
// Per lane this is two contiguous 16B chunks: k in [hf*8, hf*8+8) and
// [16+hf*8, 16+hf*8+8).
DEVFN v16h load_a_f16(const half_t* __restrict__ A, int lda) {
  const int lane = threadIdx.x & 31;
  const int m = lane & 15, hf = lane >> 4;
  const half_t* row = A + (size_t)m * lda + hf * 8;
  const v8h lo = *(const v8h*)(row);
  const v8h hi = *(const v8h*)(row + 16);
  return __builtin_shufflevector(lo, hi, 0, 1, 2, 3, 4, 5, 6, 7,
                                 8, 9, 10, 11, 12, 13, 14, 15);
}

// Same fragment sourced from LDS (generic pointer into __shared__).
DEVFN v16h load_a_lds(const half_t* sA, int lda) {
  const int lane = threadIdx.x & 31;
  const int m = lane & 15, hf = lane >> 4;
  const half_t* row = sA + m * lda + hf * 8;
  const v8h lo = *(const v8h*)(row);
  const v8h hi = *(const v8h*)(row + 16);
  return __builtin_shufflevector(lo, hi, 0, 1, 2, 3, 4, 5, 6, 7,
                                 8, 9, 10, 11, 12, 13, 14, 15);
}

// A fragment 16x32 from row-major f32 with on-the-fly f16 conversion
DEVFN v16h load_a_f32(const float* __restrict__ A, int lda) {
  const int lane = threadIdx.x & 31;
  const int m = lane & 15, hf = lane >> 4;
  const float* row = A + (size_t)m * lda;
  v16h a;
#pragma unroll
  for (int v = 0; v < 8; ++v) {
    const int k = (v < 4 ? 2 * v : 16 + 2 * (v - 4)) + hf * 8;
    a[2 * v]     = (half_t)row[k];
    a[2 * v + 1] = (half_t)row[k + 1];
  }
  return a;
}

// B fragment 32x16 (KxN) where memory holds B transposed (N-major, K
// contiguous): Bt[n*ld + k]. Per lane two 16B chunks: k in [hf*8, hf*8+8)
// and [16+hf*8, 16+hf*8+8).
DEVFN v16h load_bT(const half_t* __restrict__ Bt, int ld) {
  const int lane = threadIdx.x & 31;
  const int n = lane & 15, hf = lane >> 4;
  const half_t* col = Bt + (size_t)n * ld + hf * 8;
  const v8h lo = *(const v8h*)(col);
  const v8h hi = *(const v8h*)(col + 16);
  return __builtin_shufflevector(lo, hi, 0, 1, 2, 3, 4, 5, 6, 7,
                                 8, 9, 10, 11, 12, 13, 14, 15);
}

// ---------------- Kernel 1: LayerNorm + f16 convert ------------------------

__global__ void ln_kernel(const float* __restrict__ x,
                          const float* __restrict__ gamma,
                          const float* __restrict__ beta,
                          half_t* __restrict__ xn) {
  __shared__ float red[256];
  const int row = blockIdx.x, tid = threadIdx.x;
  const float* xr = x + (size_t)row * Dc;
  float v[4];
  float s = 0.f;
#pragma unroll
  for (int i = 0; i < 4; ++i) { v[i] = xr[tid + i * 256]; s += v[i]; }
  red[tid] = s; __syncthreads();
  for (int off = 128; off > 0; off >>= 1) {
    if (tid < off) red[tid] += red[tid + off];
    __syncthreads();
  }
  const float mu = red[0] * (1.f / Dc);
  __syncthreads();
  s = 0.f;
#pragma unroll
  for (int i = 0; i < 4; ++i) { const float d = v[i] - mu; s += d * d; }
  red[tid] = s; __syncthreads();
  for (int off = 128; off > 0; off >>= 1) {
    if (tid < off) red[tid] += red[tid + off];
    __syncthreads();
  }
  const float rs = rsqrtf(red[0] * (1.f / Dc) + LN_EPS);
#pragma unroll
  for (int i = 0; i < 4; ++i) {
    const int c = tid + i * 256;
    xn[(size_t)row * Dc + c] = (half_t)((v[i] - mu) * rs * gamma[c] + beta[c]);
  }
}

// ---------------- Kernel 2: weight transpose + f16 convert ------------------

__global__ void transpose_cvt_kernel(const float* __restrict__ W,
                                     half_t* __restrict__ Wt) {
  __shared__ float tile[16][17];
  const int bx = blockIdx.x * 16, by = blockIdx.y * 16;
  const int tx = threadIdx.x, ty = threadIdx.y;
  tile[ty][tx] = W[(size_t)(by + ty) * Dc + bx + tx];
  __syncthreads();
  Wt[(size_t)(bx + ty) * Dc + by + tx] = (half_t)tile[tx][ty];
}

// ---------------- Kernel 3: QKV projection GEMM ----------------------------
// Y(4096x1024) = Xn @ W + b, W given transposed/f16. One wave -> 16x64 tile.
// All 8 waves of a block share one 16x1024 A strip (32 KB, contiguous):
// stage it once via async global->LDS (ASYNCcnt) and read fragments from LDS.
// B fragments are register double-buffered so global_load_b128s stay in
// flight behind the WMMA chain.

__global__ void qkv_gemm_kernel(const half_t* __restrict__ xn,
                                const half_t* __restrict__ Wt,
                                const float* __restrict__ bias,
                                half_t* __restrict__ out,
                                int transposed_store) {
  __shared__ half_t sA[16 * Dc];  // 32 KB contiguous A strip
  const int wid = blockIdx.x * 8 + (threadIdx.x >> 5);
  const int mt = wid >> 4;        // 0..255 (16-row tile) -- same for all waves
  const int nt = wid & 15;        // 0..15  (64-col tile)

  // ---- async-stage A strip: 256 threads x 8 x b128 = 32768 bytes ----
  {
    const char* gsrc = (const char*)(xn + (size_t)(mt * 16) * Dc);
    const unsigned t16 = threadIdx.x * 16u;
#pragma unroll
    for (int i = 0; i < 8; ++i) {
      const unsigned off = t16 + (unsigned)i * 4096u;
      unsigned lds_addr = (unsigned)(size_t)(&sA[0]) + off;  // addr[31:0] == LDS offset
      unsigned long long gaddr = (unsigned long long)(size_t)(gsrc + off);
      asm volatile("global_load_async_to_lds_b128 %0, %1, off"
                   :: "v"(lds_addr), "v"(gaddr) : "memory");
    }
    asm volatile("s_wait_asynccnt 0" ::: "memory");
  }
  __syncthreads();

  v8f acc[4] = {};
  v16h bcur[4], bnxt[4];
#pragma unroll
  for (int j = 0; j < 4; ++j)
    bcur[j] = load_bT(Wt + (size_t)(nt * 64 + j * 16) * Dc, Dc);

  for (int kb = 0; kb < Dc / 32; ++kb) {
    if (kb + 1 < Dc / 32) {
#pragma unroll
      for (int j = 0; j < 4; ++j)
        bnxt[j] = load_bT(Wt + (size_t)(nt * 64 + j * 16) * Dc + (kb + 1) * 32,
                          Dc);
    }
    const v16h a = load_a_lds(&sA[kb * 32], Dc);
#pragma unroll
    for (int j = 0; j < 4; ++j) acc[j] = wmma_f16(a, bcur[j], acc[j]);
#pragma unroll
    for (int j = 0; j < 4; ++j) bcur[j] = bnxt[j];
  }

  const int lane = threadIdx.x & 31;
  const int n = lane & 15, hf = lane >> 4;
#pragma unroll
  for (int j = 0; j < 4; ++j) {
    const int col = nt * 64 + j * 16 + n;
    const float bcol = bias[col];
#pragma unroll
    for (int r = 0; r < 8; ++r) {
      const int row = mt * 16 + r + hf * 8;
      const float val = acc[j][r] + bcol;
      if (!transposed_store) {
        out[(size_t)row * Dc + col] = (half_t)val;
      } else {
        const int bb = row >> 10, ss = row & 1023;  // (B, Dv, S) layout
        out[((size_t)(bb * Dc + col)) * Sc + ss] = (half_t)val;
      }
    }
  }
}

// ---------------- Kernel 4: scores = Q K^T * scale, masked ------------------
// One wave -> one 16x16 tile of (S x S) for one (b,h). Upper-triangle tiles
// skip WMMA entirely (wave-uniform branch, EXEC stays all ones).

__global__ void scores_kernel(const half_t* __restrict__ q,
                              const half_t* __restrict__ k,
                              const int* __restrict__ lens,
                              float* __restrict__ attn) {
  const int kt = blockIdx.x * 8 + (threadIdx.x >> 5);
  const int qt = blockIdx.y;
  const int bh = blockIdx.z;
  const int b = bh >> 4, h = bh & 15;
  const int lane = threadIdx.x & 31;
  const int n = lane & 15, hf = lane >> 4;
  float* out = attn + ((size_t)bh * Sc + qt * 16) * Sc + kt * 16;

  if (kt > qt) {  // fully causally masked tile
#pragma unroll
    for (int r = 0; r < 8; ++r)
      out[(size_t)(r + hf * 8) * Sc + n] = -__builtin_inff();
    return;
  }

  v8f acc = {};
#pragma unroll
  for (int kb = 0; kb < 2; ++kb) {  // K = DQ = 64 -> two k32 steps
    const v16h a = load_a_f16(
        q + (size_t)(b * Sc + qt * 16) * Dc + h * 64 + kb * 32, Dc);
    const v16h bf = load_bT(
        k + (size_t)(b * Sc + kt * 16) * Dc + h * 64 + kb * 32, Dc);
    acc = wmma_f16(a, bf, acc);
  }
  const int len = lens[b];
  const int kpos = kt * 16 + n;
#pragma unroll
  for (int r = 0; r < 8; ++r) {
    const int qpos = qt * 16 + r + hf * 8;
    const bool valid = (kpos <= qpos) && (kpos < len);
    out[(size_t)(r + hf * 8) * Sc + n] =
        valid ? acc[r] * QK_SCALE : -__builtin_inff();
  }
}

// ---------------- Kernel 5: row softmax in place ---------------------------

__global__ void softmax_kernel(float* __restrict__ attn) {
  __shared__ float red[256];
  const int tid = threadIdx.x;
  float* row = attn + (size_t)blockIdx.x * Sc;
  float v[4];
  float m = -__builtin_inff();
#pragma unroll
  for (int i = 0; i < 4; ++i) { v[i] = row[tid + i * 256]; m = fmaxf(m, v[i]); }
  red[tid] = m; __syncthreads();
  for (int off = 128; off > 0; off >>= 1) {
    if (tid < off) red[tid] = fmaxf(red[tid], red[tid + off]);
    __syncthreads();
  }
  m = red[0];
  __syncthreads();
  float s = 0.f;
#pragma unroll
  for (int i = 0; i < 4; ++i) { v[i] = __expf(v[i] - m); s += v[i]; }
  red[tid] = s; __syncthreads();
  for (int off = 128; off > 0; off >>= 1) {
    if (tid < off) red[tid] += red[tid + off];
    __syncthreads();
  }
  const float inv = 1.f / red[0];
#pragma unroll
  for (int i = 0; i < 4; ++i) row[tid + i * 256] = v[i] * inv;
}

// ---------------- Kernel 6: out = attn @ V, head-merge, residual ------------
// One wave -> 16 query rows x full DV=64 for one (b,h). K-loop stops at the
// causal diagonal (probs beyond it are exactly zero).

__global__ void av_kernel(const float* __restrict__ attn,
                          const half_t* __restrict__ vt,
                          const float* __restrict__ x,
                          float* __restrict__ seq) {
  const int qt = blockIdx.x * 8 + (threadIdx.x >> 5);
  const int bh = blockIdx.y;
  const int b = bh >> 4, h = bh & 15;
  v8f acc[4] = {};
  const int kmax = qt * 16 + 16;
  for (int k0 = 0; k0 < kmax; k0 += 32) {
    const v16h a = load_a_f32(attn + ((size_t)bh * Sc + qt * 16) * Sc + k0, Sc);
#pragma unroll
    for (int j = 0; j < 4; ++j) {
      const v16h bf = load_bT(
          vt + (size_t)(b * Dc + h * 64 + j * 16) * Sc + k0, Sc);
      acc[j] = wmma_f16(a, bf, acc[j]);
    }
  }
  const int lane = threadIdx.x & 31;
  const int n = lane & 15, hf = lane >> 4;
#pragma unroll
  for (int j = 0; j < 4; ++j) {
#pragma unroll
    for (int r = 0; r < 8; ++r) {
      const int qpos = qt * 16 + r + hf * 8;
      const int col = h * 64 + j * 16 + n;
      const size_t idx = ((size_t)b * Sc + qpos) * Dc + col;
      seq[idx] = x[idx] + acc[j][r];
    }
  }
}

// ---------------- Host-side launch -----------------------------------------

extern "C" void kernel_launch(void* const* d_in, const int* in_sizes, int n_in,
                              void* d_out, int out_size, void* d_ws, size_t ws_size,
                              hipStream_t stream) {
  const float* x     = (const float*)d_in[0];   // query==key==value sequences
  const int*   lens  = (const int*)d_in[3];
  const float* gamma = (const float*)d_in[4];
  const float* beta  = (const float*)d_in[5];
  const float* Wq    = (const float*)d_in[6];
  const float* bq    = (const float*)d_in[7];
  const float* Wk    = (const float*)d_in[8];
  const float* bk    = (const float*)d_in[9];
  const float* Wv    = (const float*)d_in[10];
  const float* bv    = (const float*)d_in[11];

  float* seq  = (float*)d_out;                               // (B,S,D)
  float* attn = (float*)d_out + (size_t)Bc * Sc * Dc;        // (B,H,S,S)

  // workspace carve-up (f16)
  half_t* p = (half_t*)d_ws;
  half_t* xn  = p; p += (size_t)Mrows * Dc;   // 4096x1024
  half_t* wqt = p; p += (size_t)Dc * Dc;      // 1024x1024 transposed
  half_t* wkt = p; p += (size_t)Dc * Dc;
  half_t* wvt = p; p += (size_t)Dc * Dc;
  half_t* qh  = p; p += (size_t)Mrows * Dc;   // (B*S, H*DQ)
  half_t* kh  = p; p += (size_t)Mrows * Dc;
  half_t* vth = p; p += (size_t)Mrows * Dc;   // (B, Dv, S)

  // 1) LayerNorm -> f16 activations
  ln_kernel<<<Mrows, 256, 0, stream>>>(x, gamma, beta, xn);

  // 2) Transpose + convert weights
  dim3 tb(16, 16), tg(Dc / 16, Dc / 16);
  transpose_cvt_kernel<<<tg, tb, 0, stream>>>(Wq, wqt);
  transpose_cvt_kernel<<<tg, tb, 0, stream>>>(Wk, wkt);
  transpose_cvt_kernel<<<tg, tb, 0, stream>>>(Wv, wvt);

  // 3) Q/K/V projections (WMMA + async LDS staging); V stored transposed
  qkv_gemm_kernel<<<512, 256, 0, stream>>>(xn, wqt, bq, qh, 0);
  qkv_gemm_kernel<<<512, 256, 0, stream>>>(xn, wkt, bk, kh, 0);
  qkv_gemm_kernel<<<512, 256, 0, stream>>>(xn, wvt, bv, vth, 1);

  // 4) masked scores (WMMA) straight into d_out attn region
  scores_kernel<<<dim3(Sc / 16 / 8, Sc / 16, Bc * Hc), 256, 0, stream>>>(
      qh, kh, lens, attn);

  // 5) softmax in place (final attn output)
  softmax_kernel<<<Bc * Hc * Sc, 256, 0, stream>>>(attn);

  // 6) attn @ V + residual (WMMA, causally bounded K-loop)
  av_kernel<<<dim3(Sc / 16 / 8, Bc * Hc), 256, 0, stream>>>(attn, vth, x, seq);
}
